// SelfAttention2d_9569187135756
// MI455X (gfx1250) — matrix operand order repaired and checked by reading, not compiled, not run on hardware
//
#include <hip/hip_runtime.h>

// ---------------------------------------------------------------------------
// MI455X (gfx1250) self-attention-over-channels pipeline.
// Heavy math: v_wmma_f32_16x16x32_bf16 (wave32), fp32 accumulation.
// Staging:    global_load_async_to_lds_b128 (inline asm, ASYNCcnt-tracked)
//             with double-buffered LDS ping-pong.
// ---------------------------------------------------------------------------

typedef __attribute__((ext_vector_type(16))) __bf16 v16bf;
typedef __attribute__((ext_vector_type(8)))  __bf16 bf16x8;
typedef __attribute__((ext_vector_type(8)))  float  v8f;

#define C_DIM 768
#define HW_DIM 4096
#define NBATCH 8

#if defined(__has_builtin)
#if __has_builtin(__builtin_amdgcn_s_wait_asynccnt)
#define HAVE_WAIT_BUILTIN 1
#endif
#endif

// 16-byte async DMA: global -> LDS (ASYNCcnt-tracked).
// Inline asm (toolchain-portable; ROCm's builtin has a different signature).
// LDS byte address = low 32 bits of the generic pointer (ISA 10.2 aperture).
__device__ __forceinline__ void async_copy16(__bf16* lds, const __bf16* g)
{
    asm volatile("global_load_async_to_lds_b128 %0, %1, off"
                 :
                 : "v"((unsigned)(size_t)lds),
                   "v"((unsigned long long)(size_t)g)
                 : "memory");
}

__device__ __forceinline__ void wait_async_group(bool more_inflight)
{
#if defined(HAVE_WAIT_BUILTIN)
    if (more_inflight) __builtin_amdgcn_s_wait_asynccnt(6);
    else               __builtin_amdgcn_s_wait_asynccnt(0);
#else
    if (more_inflight) asm volatile("s_wait_asynccnt 6" ::: "memory");
    else               asm volatile("s_wait_asynccnt 0" ::: "memory");
#endif
}

// ---------------------------------------------------------------------------
// Generic WMMA GEMM:  D = A(MxK) * B'(NxK)^T  (+ bias[m]) (+ residual)
// A, B' : bf16, row-major along K.  D: bf16 or f32.
// Block = 256 threads = 8 waves (2x4). Block tile 128x256.
// K-chunk 32, double-buffered async LDS staging. Requires K % 64 == 0.
// Each wave computes a 64x64 tile = 4x4 WMMA accumulators.
// ---------------------------------------------------------------------------
#define KP 40   // LDS row pitch in bf16 (32 data + 8 pad) -> conflict-free

template<bool OUT_BF16, bool BIAS, bool RESID>
__global__ __launch_bounds__(256) void wmma_gemm(
    const __bf16* __restrict__ A, long strideA,
    const __bf16* __restrict__ B, long strideB,
    void* __restrict__ Dv, long strideD,
    const float* __restrict__ bias,
    const float* __restrict__ resid, long strideR,
    int M, int N, int K)
{
    __shared__ __align__(16) __bf16 As[2][128 * KP];   // 2 x 10 KB
    __shared__ __align__(16) __bf16 Bs[2][256 * KP];   // 2 x 20 KB

    const int tid  = threadIdx.x;
    const int lane = tid & 31;
    const int wave = tid >> 5;
    const int wm   = wave >> 2;            // 0..1  (M direction)
    const int wn   = wave & 3;             // 0..3  (N direction)
    const int m0   = blockIdx.y * 128;
    const int n0   = blockIdx.x * 256;
    const int bz   = blockIdx.z;

    const __bf16* Ab = A + (long)bz * strideA;
    const __bf16* Bb = B + (long)bz * strideB;

    v8f acc[4][4];
    #pragma unroll
    for (int i = 0; i < 4; ++i)
        #pragma unroll
        for (int j = 0; j < 4; ++j)
            #pragma unroll
            for (int e = 0; e < 8; ++e)
                acc[i][j][e] = 0.0f;

    // issue one 32-wide K-chunk into LDS buffer `buf` (6 async b128 per thread)
    auto stage = [&](int buf, int kk) {
        // A tile: 128 rows x 32 cols = 512 xfers, 2 per thread
        #pragma unroll
        for (int i = 0; i < 2; ++i) {
            const int idx = tid + i * 256;
            const int r = idx >> 2, c8 = idx & 3;
            async_copy16(&As[buf][r * KP + c8 * 8],
                         Ab + (long)(m0 + r) * K + kk + c8 * 8);
        }
        // B tile: 256 rows x 32 cols = 1024 xfers, 4 per thread
        #pragma unroll
        for (int i = 0; i < 4; ++i) {
            const int idx = tid + i * 256;
            const int r = idx >> 2, c8 = idx & 3;
            async_copy16(&Bs[buf][r * KP + c8 * 8],
                         Bb + (long)(n0 + r) * K + kk + c8 * 8);
        }
    };

    // one 32-K WMMA step out of LDS buffer `buf`
    auto compute = [&](int buf) {
        v16bf af[4], bfr[4];
        const int ml  = lane & 15;
        const int kbA = (lane >> 4) * 8;    // A frag: lanes16-31 hold K+8
        const int kbB = (lane >> 4) * 8;    // B frag: lanes16-31 hold K+8

        // A fragment (16x32 bf16), ISA 7.12.2: e<8 -> K=kbA+e ; e>=8 -> K=kbA+8+e
        #pragma unroll
        for (int mt = 0; mt < 4; ++mt) {
            const __bf16* p = &As[buf][(wm * 64 + mt * 16 + ml) * KP + kbA];
            bf16x8 lo = *(const bf16x8*)p;
            bf16x8 hi = *(const bf16x8*)(p + 16);
            v16bf a;
            #pragma unroll
            for (int e = 0; e < 8; ++e) { a[e] = lo[e]; a[e + 8] = hi[e]; }
            af[mt] = a;
        }
        // B fragment (32x16 bf16): lane holds column n=lane&15, e<8 -> K=kbB+e ; e>=8 -> K=kbB+8+e
        #pragma unroll
        for (int nt = 0; nt < 4; ++nt) {
            const __bf16* p = &Bs[buf][(wn * 64 + nt * 16 + ml) * KP + kbB];
            bf16x8 lo = *(const bf16x8*)p;
            bf16x8 hi = *(const bf16x8*)(p + 16);
            v16bf bv;
            #pragma unroll
            for (int e = 0; e < 8; ++e) { bv[e] = lo[e]; bv[e + 8] = hi[e]; }
            bfr[nt] = bv;
        }
        #pragma unroll
        for (int mt = 0; mt < 4; ++mt)
            #pragma unroll
            for (int nt = 0; nt < 4; ++nt)
                acc[mt][nt] = __builtin_amdgcn_wmma_f32_16x16x32_bf16(
                    false, af[mt], false, bfr[nt],
                    (short)0, acc[mt][nt], false, false);
    };

    // ---- ping-pong main loop (buffer indices compile-time constant) --------
    stage(0, 0);
    for (int k0 = 0; k0 < K; k0 += 64) {
        const bool more1 = (k0 + 32) < K;
        if (more1) stage(1, k0 + 32);
        wait_async_group(more1);     // chunk in buf0 complete (this wave)
        __syncthreads();             // ... and for all waves
        compute(0);
        __syncthreads();             // buf0 free for overwrite

        const bool more2 = (k0 + 64) < K;
        if (more2) stage(0, k0 + 64);
        wait_async_group(more2);     // chunk in buf1 complete
        __syncthreads();
        compute(1);
        __syncthreads();
    }

    // ---- store: C/D layout -> lane n = lane&15, VGPR r -> M = r + 8*(lane>=16)
    const int nl = lane & 15;
    const int mo = (lane >> 4) * 8;
    const float* rb = RESID ? (resid + (long)bz * strideR) : nullptr;
    #pragma unroll
    for (int mt = 0; mt < 4; ++mt) {
        #pragma unroll
        for (int nt = 0; nt < 4; ++nt) {
            const int gm = m0 + wm * 64 + mt * 16 + mo;
            const int gn = n0 + wn * 64 + nt * 16 + nl;
            #pragma unroll
            for (int r = 0; r < 8; ++r) {
                float val = acc[mt][nt][r];
                const int row = gm + r;
                if (BIAS) val += bias[row];
                const long off = (long)row * N + gn;
                if (RESID) val += rb[off];
                if (OUT_BF16)
                    ((__bf16*)Dv + (long)bz * strideD)[off] = (__bf16)val;
                else
                    ((float*)Dv + (long)bz * strideD)[off] = val;
            }
        }
    }
}

// ---------------------------------------------------------------------------
// fp32 -> bf16 elementwise cast (for the 768x768 weight matrices)
// ---------------------------------------------------------------------------
__global__ __launch_bounds__(256) void cast_f32_bf16(
    const float* __restrict__ in, __bf16* __restrict__ out, long n)
{
    const long i = (long)blockIdx.x * 256 + threadIdx.x;
    if (i < n) out[i] = (__bf16)in[i];
}

// ---------------------------------------------------------------------------
// Tiled transpose (R x Cc) -> (Cc x R), output bf16, per batch.
// ---------------------------------------------------------------------------
template<typename Tin>
__global__ __launch_bounds__(256) void transpose_to_bf16(
    const Tin* __restrict__ in, __bf16* __restrict__ out, int R, int Cc)
{
    __shared__ __bf16 tile[32][33];
    const int b = blockIdx.z;
    in  += (long)b * R * Cc;
    out += (long)b * R * Cc;
    const int c0 = blockIdx.x * 32;
    const int r0 = blockIdx.y * 32;
    const int tx = threadIdx.x, ty = threadIdx.y;   // 32 x 8
    #pragma unroll
    for (int i = 0; i < 32; i += 8)
        tile[ty + i][tx] = (__bf16)(float)in[(long)(r0 + ty + i) * Cc + c0 + tx];
    __syncthreads();
    #pragma unroll
    for (int i = 0; i < 32; i += 8)
        out[(long)(c0 + ty + i) * R + r0 + tx] = tile[tx][ty + i];
}

// ---------------------------------------------------------------------------
// Row softmax over 768 logits with folded 1/hw scale; bf16 output.
// One 256-thread block (8 wave32) per row.
// ---------------------------------------------------------------------------
__global__ __launch_bounds__(256) void softmax_row(
    const float* __restrict__ logits, __bf16* __restrict__ att)
{
    const int b   = blockIdx.y;
    const int row = blockIdx.x;
    const float* lp = logits + ((long)b * C_DIM + row) * C_DIM;
    __bf16*      ap = att    + ((long)b * C_DIM + row) * C_DIM;
    const float scale = 1.0f / (float)HW_DIM;   // (hw^-0.5)^2 on q and k

    const int lane = threadIdx.x & 31;
    const int wv   = threadIdx.x >> 5;
    __shared__ float red[8];

    float v[3];
    float mx = -1e30f;
    #pragma unroll
    for (int i = 0; i < 3; ++i) {
        v[i] = lp[threadIdx.x + i * 256] * scale;
        mx = fmaxf(mx, v[i]);
    }
    #pragma unroll
    for (int off = 16; off; off >>= 1) mx = fmaxf(mx, __shfl_xor(mx, off, 32));
    if (lane == 0) red[wv] = mx;
    __syncthreads();
    if (wv == 0) {
        float m2 = (lane < 8) ? red[lane] : -1e30f;
        #pragma unroll
        for (int off = 4; off; off >>= 1) m2 = fmaxf(m2, __shfl_xor(m2, off, 32));
        if (lane == 0) red[0] = m2;
    }
    __syncthreads();
    mx = red[0];

    float s = 0.0f;
    #pragma unroll
    for (int i = 0; i < 3; ++i) { v[i] = __expf(v[i] - mx); s += v[i]; }
    #pragma unroll
    for (int off = 16; off; off >>= 1) s += __shfl_xor(s, off, 32);
    __syncthreads();
    if (lane == 0) red[wv] = s;
    __syncthreads();
    if (wv == 0) {
        float s2 = (lane < 8) ? red[lane] : 0.0f;
        #pragma unroll
        for (int off = 4; off; off >>= 1) s2 += __shfl_xor(s2, off, 32);
        if (lane == 0) red[0] = s2;
    }
    __syncthreads();
    const float inv = 1.0f / red[0];
    #pragma unroll
    for (int i = 0; i < 3; ++i)
        ap[threadIdx.x + i * 256] = (__bf16)(v[i] * inv);
}

// ---------------------------------------------------------------------------
// Source's transpose(1,2).view(n,c,h,w) reinterpret + build (hw,c) B-operand:
//   Z[j][c'] = y[(c'*hw + j) % c][(c'*hw + j) / c]
// y (bf16, c x hw, 50MB) is L2-resident (192MB) so scattered reads are cheap.
// ---------------------------------------------------------------------------
__global__ __launch_bounds__(256) void permute_y(
    const __bf16* __restrict__ y, __bf16* __restrict__ Z)
{
    const int b = blockIdx.z;
    const __bf16* yb = y + (long)b * C_DIM * HW_DIM;
    __bf16*       Zb = Z + (long)b * C_DIM * HW_DIM;
    const int j  = blockIdx.y;
    const int cc = blockIdx.x * 256 + threadIdx.x;
    const long m = (long)cc * HW_DIM + j;
    const int d  = (int)(m % C_DIM);
    const int i  = (int)(m / C_DIM);
    Zb[(long)j * C_DIM + cc] = yb[(long)d * HW_DIM + i];
}

// ---------------------------------------------------------------------------
// Host launcher
// ---------------------------------------------------------------------------
extern "C" void kernel_launch(void* const* d_in, const int* in_sizes, int n_in,
                              void* d_out, int out_size, void* d_ws, size_t ws_size,
                              hipStream_t stream)
{
    const float* x  = (const float*)d_in[0];
    const float* Wq = (const float*)d_in[1];
    const float* bq = (const float*)d_in[2];
    const float* Wk = (const float*)d_in[3];
    const float* bk = (const float*)d_in[4];
    const float* Wv = (const float*)d_in[5];
    const float* bv = (const float*)d_in[6];
    const float* Wo = (const float*)d_in[7];
    const float* bo = (const float*)d_in[8];

    const long CD = C_DIM, HW = HW_DIM, NB = NBATCH;
    const long wN   = CD * CD;            // 589,824 weight elems
    const long big  = NB * CD * HW;       // 25,165,824 elems

    char* p = (char*)d_ws;
    auto alloc = [&](size_t bytes) -> char* {
        char* r = p; p += (bytes + 255) & ~(size_t)255; return r;
    };
    __bf16* wqb  = (__bf16*)alloc(wN * 2);
    __bf16* wkb  = (__bf16*)alloc(wN * 2);
    __bf16* wvb  = (__bf16*)alloc(wN * 2);
    __bf16* wob  = (__bf16*)alloc(wN * 2);
    __bf16* xT   = (__bf16*)alloc(big * 2);   // (n, hw, c) ; reused later for y
    __bf16* qb   = (__bf16*)alloc(big * 2);   // (n, c, hw) ; reused later for Z
    __bf16* kb   = (__bf16*)alloc(big * 2);   // (n, c, hw)
    __bf16* vb   = (__bf16*)alloc(big * 2);   // (n, c, hw)
    __bf16* vT   = (__bf16*)alloc(big * 2);   // (n, hw, c)
    float*  logt = (float*) alloc(NB * CD * CD * 4);
    __bf16* att  = (__bf16*)alloc(NB * CD * CD * 2);
    __bf16* yb   = xT;                        // xT dead after projections
    __bf16* Zb   = qb;                        // q dead after logits

    // 1) weights -> bf16
    cast_f32_bf16<<<dim3((wN + 255) / 256), dim3(256), 0, stream>>>(Wq, wqb, wN);
    cast_f32_bf16<<<dim3((wN + 255) / 256), dim3(256), 0, stream>>>(Wk, wkb, wN);
    cast_f32_bf16<<<dim3((wN + 255) / 256), dim3(256), 0, stream>>>(Wv, wvb, wN);
    cast_f32_bf16<<<dim3((wN + 255) / 256), dim3(256), 0, stream>>>(Wo, wob, wN);

    // 2) x (n,c,hw) f32 -> xT (n,hw,c) bf16
    dim3 tb(32, 8, 1);
    transpose_to_bf16<float><<<dim3(HW / 32, CD / 32, NB), tb, 0, stream>>>(
        x, xT, (int)CD, (int)HW);

    // 3) projections: q/k/v = W @ x + b     (M=768, N=4096, K=768)
    dim3 gProj(HW / 256, CD / 128, NB);
    wmma_gemm<true, true, false><<<gProj, dim3(256), 0, stream>>>(
        wqb, 0, xT, HW * CD, qb, CD * HW, bq, nullptr, 0, (int)CD, (int)HW, (int)CD);
    wmma_gemm<true, true, false><<<gProj, dim3(256), 0, stream>>>(
        wkb, 0, xT, HW * CD, kb, CD * HW, bk, nullptr, 0, (int)CD, (int)HW, (int)CD);
    wmma_gemm<true, true, false><<<gProj, dim3(256), 0, stream>>>(
        wvb, 0, xT, HW * CD, vb, CD * HW, bv, nullptr, 0, (int)CD, (int)HW, (int)CD);

    // 4) v (n,c,hw) -> vT (n,hw,c)
    transpose_to_bf16<__bf16><<<dim3(HW / 32, CD / 32, NB), tb, 0, stream>>>(
        vb, vT, (int)CD, (int)HW);

    // 5) logits = q @ k^T                   (M=768, N=768, K=4096)
    dim3 gAtt(CD / 256, CD / 128, NB);
    wmma_gemm<false, false, false><<<gAtt, dim3(256), 0, stream>>>(
        qb, CD * HW, kb, CD * HW, logt, CD * CD, nullptr, nullptr, 0,
        (int)CD, (int)CD, (int)HW);

    // 6) softmax rows (scale 1/hw folded in), bf16 att
    softmax_row<<<dim3(CD, NB), dim3(256), 0, stream>>>(logt, att);

    // 7) y = att @ v                        (M=768, N=4096, K=768)
    wmma_gemm<true, false, false><<<gProj, dim3(256), 0, stream>>>(
        att, CD * CD, vT, HW * CD, yb, CD * HW, nullptr, nullptr, 0,
        (int)CD, (int)HW, (int)CD);

    // 8) the source's transpose+view reinterpret, emitted as (hw,c) operand
    permute_y<<<dim3(CD / 256, HW, NB), dim3(256), 0, stream>>>(yb, Zb);

    // 9) out = x + Wo @ y_reinterpret + bo  (M=768, N=4096, K=768)
    wmma_gemm<false, true, true><<<gProj, dim3(256), 0, stream>>>(
        wob, 0, Zb, HW * CD, (float*)d_out, CD * HW, bo, x, CD * HW,
        (int)CD, (int)HW, (int)CD);

    (void)in_sizes; (void)n_in; (void)out_size; (void)ws_size;
}
